// NonLocalBlock_83691732730068
// MI455X (gfx1250) — matrix operand order repaired and checked by reading, not compiled, not run on hardware
//
#include <hip/hip_runtime.h>
#include <hip/hip_bf16.h>

typedef _Float16 v16h __attribute__((ext_vector_type(16)));
typedef float    v8f  __attribute__((ext_vector_type(8)));

#define N_TOK 2048
#define C_IN  1024
#define HID   64
#define BATCH 8

union Frag16 { v16h v; unsigned u[8]; _Float16 h[16]; };

__device__ __forceinline__ unsigned pack2f(float a, float b) {
    union { _Float16 h[2]; unsigned u; } p;
    p.h[0] = (_Float16)a; p.h[1] = (_Float16)b;
    return p.u;
}
// A fragment: lane holds row m=L%16; VGPR j holds K pair starting at:
__device__ __forceinline__ int k0A(int j, int half) { return 2*j + ((j >= 4) ? 8 : 0) + 8*half; }
// B fragment: lane holds col n=L%16; VGPR j holds K pair starting at:
__device__ __forceinline__ int k0B(int j, int half) { return 2*j + ((j >= 4) ? 8 : 0) + 8*half; }

__device__ __forceinline__ v8f wmma_f16(v16h a, v16h b, v8f c) {
    return __builtin_amdgcn_wmma_f32_16x16x32_f16(false, a, false, b, (short)0, c, false, false);
}

// ---------------------------------------------------------------------------
// Zero the BN accumulators (workspace is poisoned once, never re-poisoned).
__global__ void k_zero(float* __restrict__ p, int n) {
    int i = blockIdx.x * blockDim.x + threadIdx.x;
    if (i < n) p[i] = 0.0f;
}

// ---------------------------------------------------------------------------
// Projections: P[o,n] = sum_c W[o,c]*x[b,c,n] + bias[o]  (o=64, n=2048, K=1024)
// theta/phi stored [B][n][64] f16 ; g stored [B][64][n] f16.
// Block = 256 thr = 8 waves: 4 waves over o (64), 2 over n (128 cols/block).
__global__ __launch_bounds__(256) void k_proj(
    const float* __restrict__ x,
    const float* __restrict__ w0, const float* __restrict__ b0,
    const float* __restrict__ w1, const float* __restrict__ b1,
    const float* __restrict__ w2, const float* __restrict__ b2,
    _Float16* __restrict__ theta, _Float16* __restrict__ phi,
    _Float16* __restrict__ g)
{
    __shared__ _Float16 sm[64*34 + 128*34];
    _Float16* Wl = sm;           // [64][34]  A tile (o,k)
    _Float16* Xl = sm + 64*34;   // [128][34] B tile transposed (n,k)

    const int proj = blockIdx.y;
    const int b    = blockIdx.z;
    const int n0   = blockIdx.x * 128;
    const float* Wm = (proj == 0) ? w0 : (proj == 1) ? w1 : w2;
    const float* bv = (proj == 0) ? b0 : (proj == 1) ? b1 : b2;
    const float* xb = x + (size_t)b * C_IN * N_TOK;

    const int tid  = threadIdx.x;
    const int lane = tid & 31;
    const int wid  = tid >> 5;
    const int wo   = wid & 3;     // o-tile index (0..3)
    const int wn   = wid >> 2;    // n-half (0..1)
    const int lm   = lane & 15;
    const int half = lane >> 4;

    v8f acc[4] = {};

    for (int kk = 0; kk < C_IN; kk += 32) {
        { // weights 64x32 -> f16 LDS
            int o  = tid >> 2;
            int kq = (tid & 3) * 8;
            const float* src = Wm + (size_t)o * C_IN + kk + kq;
            float4 f0 = *(const float4*)(src);
            float4 f1 = *(const float4*)(src + 4);
            unsigned* dst = (unsigned*)(Wl + o*34 + kq);
            dst[0] = pack2f(f0.x, f0.y); dst[1] = pack2f(f0.z, f0.w);
            dst[2] = pack2f(f1.x, f1.y); dst[3] = pack2f(f1.z, f1.w);
        }
        { // x 32x128 -> transposed f16 LDS
            int kr = tid >> 3;
            int nq = (tid & 7) * 16;
            const float* src = xb + (size_t)(kk + kr) * N_TOK + n0 + nq;
            if (kk + 32 < C_IN) __builtin_prefetch(src + 32 * N_TOK, 0, 1);
#pragma unroll
            for (int q = 0; q < 4; ++q) {
                float4 f = *(const float4*)(src + q*4);
                Xl[(nq + q*4 + 0)*34 + kr] = (_Float16)f.x;
                Xl[(nq + q*4 + 1)*34 + kr] = (_Float16)f.y;
                Xl[(nq + q*4 + 2)*34 + kr] = (_Float16)f.z;
                Xl[(nq + q*4 + 3)*34 + kr] = (_Float16)f.w;
            }
        }
        __syncthreads();

        Frag16 a;
#pragma unroll
        for (int j = 0; j < 8; ++j)
            a.u[j] = *(const unsigned*)(Wl + (wo*16 + lm)*34 + k0A(j, half));
#pragma unroll
        for (int t = 0; t < 4; ++t) {
            Frag16 bb;
#pragma unroll
            for (int j = 0; j < 8; ++j)
                bb.u[j] = *(const unsigned*)(Xl + (wn*64 + t*16 + lm)*34 + k0B(j, half));
            acc[t] = wmma_f16(a.v, bb.v, acc[t]);
        }
        __syncthreads();
    }

    float bias[8];
#pragma unroll
    for (int r = 0; r < 8; ++r) bias[r] = bv[wo*16 + 8*half + r];

    if (proj < 2) {                       // theta/phi: [n][64], 16B packed store
        _Float16* out = ((proj == 0) ? theta : phi) + (size_t)b * N_TOK * HID;
#pragma unroll
        for (int t = 0; t < 4; ++t) {
            int gn = n0 + wn*64 + t*16 + lm;
            union { _Float16 h[8]; uint4 q; } pk;
#pragma unroll
            for (int r = 0; r < 8; ++r) pk.h[r] = (_Float16)(acc[t][r] + bias[r]);
            *(uint4*)(out + (size_t)gn * HID + wo*16 + 8*half) = pk.q;
        }
    } else {                              // g: [64][n]
        _Float16* out = g + (size_t)b * HID * N_TOK;
#pragma unroll
        for (int t = 0; t < 4; ++t) {
            int gn = n0 + wn*64 + t*16 + lm;
#pragma unroll
            for (int r = 0; r < 8; ++r)
                out[(size_t)(wo*16 + 8*half + r) * N_TOK + gn] =
                    (_Float16)(acc[t][r] + bias[r]);
        }
    }
}

// ---------------------------------------------------------------------------
// Fused y = ((theta^T phi)/N) g  without materializing the 2048x2048 f.
// Each wave owns 16 rows n; loops m in chunks of 32; S bounced through LDS
// (D-layout -> A-layout transpose), guarded by s_wait_dscnt.
__global__ __launch_bounds__(256) void k_attn(
    const _Float16* __restrict__ theta, const _Float16* __restrict__ phi,
    const _Float16* __restrict__ g, _Float16* __restrict__ y)
{
    __shared__ _Float16 sm[8 * 16 * 34];
    const int tid = threadIdx.x, lane = tid & 31, wid = tid >> 5;
    const int lm = lane & 15, half = lane >> 4;
    const int b  = blockIdx.y;
    const int nb = blockIdx.x * 128 + wid * 16;
    const _Float16* th = theta + (size_t)b * N_TOK * HID;
    const _Float16* ph = phi   + (size_t)b * N_TOK * HID;
    const _Float16* gd = g     + (size_t)b * HID * N_TOK;
    _Float16* Sl = sm + wid * 16 * 34;

    Frag16 aTh[2];
#pragma unroll
    for (int kc = 0; kc < 2; ++kc)
#pragma unroll
        for (int j = 0; j < 8; ++j)
            aTh[kc].u[j] = *(const unsigned*)(th + (size_t)(nb + lm)*HID + kc*32 + k0A(j, half));

    v8f accY[4] = {};
    const float invN = 1.0f / (float)N_TOK;

    for (int m0 = 0; m0 < N_TOK; m0 += 32) {
#pragma unroll
        for (int st = 0; st < 2; ++st) {      // S chunk 16(n) x 32(m)
            v8f s = {};
            const int m = m0 + st*16 + lm;
#pragma unroll
            for (int kc = 0; kc < 2; ++kc) {  // K over hid=64
                Frag16 bp;
#pragma unroll
                for (int j = 0; j < 8; ++j)
                    bp.u[j] = *(const unsigned*)(ph + (size_t)m*HID + kc*32 + k0B(j, half));
                s = wmma_f16(aTh[kc].v, bp.v, s);
            }
#pragma unroll
            for (int r = 0; r < 8; ++r)
                Sl[(r + 8*half)*34 + st*16 + lm] = (_Float16)(s[r] * invN);
        }
        asm volatile("s_wait_dscnt 0" ::: "memory");
        Frag16 a2;
#pragma unroll
        for (int j = 0; j < 8; ++j)
            a2.u[j] = *(const unsigned*)(Sl + lm*34 + k0A(j, half));
#pragma unroll
        for (int dt = 0; dt < 4; ++dt) {      // Y(16x64) += S(16x32) * g(32x64)
            Frag16 bg;
#pragma unroll
            for (int j = 0; j < 8; ++j)
                bg.u[j] = *(const unsigned*)(gd + (size_t)(dt*16 + lm)*N_TOK + m0 + k0B(j, half));
            accY[dt] = wmma_f16(a2.v, bg.v, accY[dt]);
        }
        asm volatile("s_wait_dscnt 0" ::: "memory");
    }

    _Float16* yb = y + (size_t)b * N_TOK * HID;  // y stored [n][64]
#pragma unroll
    for (int dt = 0; dt < 4; ++dt)
#pragma unroll
        for (int r = 0; r < 8; ++r)
            yb[(size_t)(nb + r + 8*half)*HID + dt*16 + lm] = (_Float16)accY[dt][r];
}

// ---------------------------------------------------------------------------
// wy = Ww(1024x64) * y^T + wb ; accumulate per-channel sum / sumsq (atomics).
__global__ __launch_bounds__(256) void k_wy_stats(
    const _Float16* __restrict__ y, const float* __restrict__ Ww,
    const float* __restrict__ wb, float* __restrict__ gsum, float* __restrict__ gsq)
{
    const int tid = threadIdx.x, lane = tid & 31, wid = tid >> 5;
    const int lm = lane & 15, half = lane >> 4;
    const int b  = blockIdx.z;
    const int cb = blockIdx.y * 128 + wid * 16;
    const int n0 = blockIdx.x * 128;
    const _Float16* yb = y + (size_t)b * N_TOK * HID;

    Frag16 aW[2];
#pragma unroll
    for (int kc = 0; kc < 2; ++kc)
#pragma unroll
        for (int j = 0; j < 8; ++j) {
            const float* p = Ww + (size_t)(cb + lm)*HID + kc*32 + k0A(j, half);
            aW[kc].u[j] = pack2f(p[0], p[1]);
        }
    float bias[8];
#pragma unroll
    for (int r = 0; r < 8; ++r) bias[r] = wb[cb + 8*half + r];

    float ps[8] = {0,0,0,0,0,0,0,0};
    float pq[8] = {0,0,0,0,0,0,0,0};
#pragma unroll
    for (int t = 0; t < 8; ++t) {
        v8f acc;
#pragma unroll
        for (int r = 0; r < 8; ++r) acc[r] = bias[r];
        const int n = n0 + t*16 + lm;
#pragma unroll
        for (int kc = 0; kc < 2; ++kc) {
            Frag16 by;
#pragma unroll
            for (int j = 0; j < 8; ++j)
                by.u[j] = *(const unsigned*)(yb + (size_t)n*HID + kc*32 + k0B(j, half));
            acc = wmma_f16(aW[kc].v, by.v, acc);
        }
#pragma unroll
        for (int r = 0; r < 8; ++r) { ps[r] += acc[r]; pq[r] += acc[r]*acc[r]; }
    }
#pragma unroll
    for (int r = 0; r < 8; ++r) {
        float v = ps[r], q = pq[r];
#pragma unroll
        for (int off = 8; off >= 1; off >>= 1) {
            v += __shfl_xor(v, off, 32);
            q += __shfl_xor(q, off, 32);
        }
        if (lm == 0) {
            int c = cb + 8*half + r;
            atomicAdd(&gsum[c], v);
            atomicAdd(&gsq[c],  q);
        }
    }
}

// ---------------------------------------------------------------------------
__global__ void k_finalize(const float* __restrict__ gsum, const float* __restrict__ gsq,
                           const float* __restrict__ gamma, const float* __restrict__ beta,
                           float* __restrict__ scale, float* __restrict__ shift)
{
    int c = blockIdx.x * blockDim.x + threadIdx.x;
    if (c < C_IN) {
        const float cnt = (float)((size_t)BATCH * N_TOK);
        float mean = gsum[c] / cnt;
        float var  = gsq[c] / cnt - mean * mean;
        float inv  = rsqrtf(var + 1e-5f);
        float sc   = gamma[c] * inv;
        scale[c] = sc;
        shift[c] = beta[c] - mean * sc;
    }
}

// ---------------------------------------------------------------------------
// Recompute wy (cheap), apply BN affine + residual x, write out.
__global__ __launch_bounds__(256) void k_wy_out(
    const _Float16* __restrict__ y, const float* __restrict__ Ww,
    const float* __restrict__ wb, const float* __restrict__ scale,
    const float* __restrict__ shift, const float* __restrict__ x,
    float* __restrict__ out)
{
    const int tid = threadIdx.x, lane = tid & 31, wid = tid >> 5;
    const int lm = lane & 15, half = lane >> 4;
    const int b  = blockIdx.z;
    const int cb = blockIdx.y * 128 + wid * 16;
    const int n0 = blockIdx.x * 128;
    const _Float16* yb = y + (size_t)b * N_TOK * HID;

    Frag16 aW[2];
#pragma unroll
    for (int kc = 0; kc < 2; ++kc)
#pragma unroll
        for (int j = 0; j < 8; ++j) {
            const float* p = Ww + (size_t)(cb + lm)*HID + kc*32 + k0A(j, half);
            aW[kc].u[j] = pack2f(p[0], p[1]);
        }
    float bias[8], sc[8], sh[8];
#pragma unroll
    for (int r = 0; r < 8; ++r) {
        int c = cb + 8*half + r;
        bias[r] = wb[c]; sc[r] = scale[c]; sh[r] = shift[c];
    }

#pragma unroll
    for (int t = 0; t < 8; ++t) {
        v8f acc;
#pragma unroll
        for (int r = 0; r < 8; ++r) acc[r] = bias[r];
        const int n = n0 + t*16 + lm;
#pragma unroll
        for (int kc = 0; kc < 2; ++kc) {
            Frag16 by;
#pragma unroll
            for (int j = 0; j < 8; ++j)
                by.u[j] = *(const unsigned*)(yb + (size_t)n*HID + kc*32 + k0B(j, half));
            acc = wmma_f16(aW[kc].v, by.v, acc);
        }
#pragma unroll
        for (int r = 0; r < 8; ++r) {
            size_t idx = ((size_t)b * C_IN + (cb + 8*half + r)) * N_TOK + n;
            out[idx] = acc[r] * sc[r] + sh[r] + x[idx];
        }
    }
}

// ---------------------------------------------------------------------------
extern "C" void kernel_launch(void* const* d_in, const int* in_sizes, int n_in,
                              void* d_out, int out_size, void* d_ws, size_t ws_size,
                              hipStream_t stream)
{
    (void)in_sizes; (void)n_in; (void)out_size; (void)ws_size;
    const float* x     = (const float*)d_in[0];
    const float* tw    = (const float*)d_in[1];
    const float* tb    = (const float*)d_in[2];
    const float* pw    = (const float*)d_in[3];
    const float* pb    = (const float*)d_in[4];
    const float* gw    = (const float*)d_in[5];
    const float* gb    = (const float*)d_in[6];
    const float* ww    = (const float*)d_in[7];
    const float* wbv   = (const float*)d_in[8];
    const float* gamma = (const float*)d_in[9];
    const float* beta  = (const float*)d_in[10];
    float* out = (float*)d_out;

    char* ws = (char*)d_ws;
    const size_t projBytes = (size_t)BATCH * N_TOK * HID * sizeof(_Float16); // 2 MB each
    _Float16* theta = (_Float16*)(ws);
    _Float16* phi   = (_Float16*)(ws + 1*projBytes);
    _Float16* g     = (_Float16*)(ws + 2*projBytes);
    _Float16* y     = (_Float16*)(ws + 3*projBytes);
    float* gsum  = (float*)(ws + 4*projBytes);
    float* gsq   = gsum + C_IN;
    float* scale = gsq + C_IN;
    float* shift = scale + C_IN;

    k_zero<<<dim3((2*C_IN + 255)/256), 256, 0, stream>>>(gsum, 2*C_IN);
    k_proj<<<dim3(N_TOK/128, 3, BATCH), 256, 0, stream>>>(
        x, tw, tb, pw, pb, gw, gb, theta, phi, g);
    k_attn<<<dim3(N_TOK/128, BATCH), 256, 0, stream>>>(theta, phi, g, y);
    k_wy_stats<<<dim3(N_TOK/128, C_IN/128, BATCH), 256, 0, stream>>>(
        y, ww, wbv, gsum, gsq);
    k_finalize<<<dim3((C_IN + 255)/256), 256, 0, stream>>>(
        gsum, gsq, gamma, beta, scale, shift);
    k_wy_out<<<dim3(N_TOK/128, C_IN/128, BATCH), 256, 0, stream>>>(
        y, ww, wbv, scale, shift, x, out);
}
